// OCR_10282151707349
// MI455X (gfx1250) — matrix operand order repaired and checked by reading, not compiled, not run on hardware
//
#include <hip/hip_runtime.h>
#include <hip/hip_bf16.h>

#define ED   320
#define NH   4
#define HD   80
#define NL   5
#define FF   2048
#define NHYP 64
#define TC   255
#define TKC  256   // keys per sequence: self (255 cached + 1) and cross (S_MEM)

typedef __attribute__((ext_vector_type(16))) __bf16 v16bf;
typedef __attribute__((ext_vector_type(8)))  __bf16 v8bf;
typedef __attribute__((ext_vector_type(8)))  float  v8f;

__device__ __forceinline__ unsigned short f2bf(float f) {
    unsigned u = __float_as_uint(f);
    unsigned r = u + 0x7FFFu + ((u >> 16) & 1u);   // round-to-nearest-even
    return (unsigned short)(r >> 16);
}
__device__ __forceinline__ float bf2f(unsigned short h) {
    return __uint_as_float(((unsigned)h) << 16);
}

// ---------------------------------------------------------------------------
// Generic bf16 WMMA GEMM:  C[M x N] = A[M x K] @ W[K x N] + bias
// A: bf16 row-major with row stride lda (elements).
// Wt: bf16, PRE-TRANSPOSED [N x K] so B fragments are contiguous per lane.
// mode: 0 = store bf16, 1 = store f32, 2 = f32 residual add (C += v)
// Each wave computes a 16x64 tile: one A fragment feeds 4 WMMAs.
// ---------------------------------------------------------------------------
__global__ __launch_bounds__(256) void gemm_bf16_wmma(
    const unsigned short* __restrict__ A, long long lda,
    const unsigned short* __restrict__ Wt,
    const float* __restrict__ bias,
    void* __restrict__ Cout,
    int M, int N, int K, int mode, int relu)
{
    const int lane = threadIdx.x & 31;
    const int wave = threadIdx.x >> 5;
    const int Mt = M >> 4;
    const int Ng = N >> 6;
    int gw = (int)blockIdx.x * 8 + wave;
    if (gw >= Mt * Ng) return;          // wave-uniform: EXEC stays all-ones
    const int mt  = gw / Ng;
    const int ng  = gw - mt * Ng;
    const int grp = lane >> 4;          // 0 or 1 (half-wave)
    const int lr  = lane & 15;

    // A fragment (16x32 bf16): lane = row M, halves 0-15 at
    // K=grp*16..grp*16+15 (contiguous, matching B).
    const __bf16* arow = (const __bf16*)A + (long long)(mt * 16 + lr) * lda + grp * 16;
    // B fragment (32x16): lanes 0-15 -> N=lane, K=0..15; lanes 16-31 -> K=16..31.
    const __bf16* b0 = (const __bf16*)Wt + (long long)(ng * 64 +  0 + lr) * K + grp * 16;
    const __bf16* b1 = (const __bf16*)Wt + (long long)(ng * 64 + 16 + lr) * K + grp * 16;
    const __bf16* b2 = (const __bf16*)Wt + (long long)(ng * 64 + 32 + lr) * K + grp * 16;
    const __bf16* b3 = (const __bf16*)Wt + (long long)(ng * 64 + 48 + lr) * K + grp * 16;

    v8f acc[4] = {};
    for (int k0 = 0; k0 < K; k0 += 32) {
        v8bf alo = *(const v8bf*)(arow + k0);
        v8bf ahi = *(const v8bf*)(arow + k0 + 8);
        v16bf af = __builtin_shufflevector(alo, ahi,
                    0,1,2,3,4,5,6,7,8,9,10,11,12,13,14,15);
        v16bf fb0 = *(const v16bf*)(b0 + k0);
        v16bf fb1 = *(const v16bf*)(b1 + k0);
        v16bf fb2 = *(const v16bf*)(b2 + k0);
        v16bf fb3 = *(const v16bf*)(b3 + k0);
        __builtin_prefetch((const void*)(arow + k0 + 32), 0, 1);
        acc[0] = __builtin_amdgcn_wmma_f32_16x16x32_bf16(false, af, false, fb0,
                                                         (short)0, acc[0], false, false);
        acc[1] = __builtin_amdgcn_wmma_f32_16x16x32_bf16(false, af, false, fb1,
                                                         (short)0, acc[1], false, false);
        acc[2] = __builtin_amdgcn_wmma_f32_16x16x32_bf16(false, af, false, fb2,
                                                         (short)0, acc[2], false, false);
        acc[3] = __builtin_amdgcn_wmma_f32_16x16x32_bf16(false, af, false, fb3,
                                                         (short)0, acc[3], false, false);
    }

    // C/D layout: VGPR r -> row = mt*16 + grp*8 + r, col = ng*64 + j*16 + lr.
#pragma unroll
    for (int j = 0; j < 4; ++j) {
        int col = ng * 64 + j * 16 + lr;
        float bv = bias ? bias[col] : 0.0f;
#pragma unroll
        for (int r = 0; r < 8; ++r) {
            int row = mt * 16 + grp * 8 + r;
            float v = acc[j][r] + bv;
            if (relu) v = fmaxf(v, 0.0f);
            long long idx = (long long)row * N + col;
            if (mode == 0)      ((unsigned short*)Cout)[idx] = f2bf(v);
            else if (mode == 1) ((float*)Cout)[idx] = v;
            else                ((float*)Cout)[idx] += v;
        }
    }
}

// ---------------------------------------------------------------------------
// Weight transpose + f32->bf16 convert: W[K x N] f32 -> Wt[N x K] bf16
// ---------------------------------------------------------------------------
__global__ __launch_bounds__(256) void transconv_kernel(
    const float* __restrict__ W, unsigned short* __restrict__ Wt, int K, int N)
{
    __shared__ float t[16][17];
    int kt = K >> 4;
    int bk = blockIdx.x % kt;
    int bn = blockIdx.x / kt;
    int tx = threadIdx.x & 15;
    int ty = threadIdx.x >> 4;
    t[ty][tx] = W[(long long)(bk * 16 + ty) * N + bn * 16 + tx];
    __syncthreads();
    Wt[(long long)(bn * 16 + ty) * K + bk * 16 + tx] = f2bf(t[tx][ty]);
}

__global__ void f32_to_bf16_kernel(const float* __restrict__ X,
                                   unsigned short* __restrict__ Y, long long n)
{
    long long i = (long long)blockIdx.x * 256 + threadIdx.x;
    if (i < n) Y[i] = f2bf(X[i]);
}

// ---------------------------------------------------------------------------
// Token embedding gather: tgt[b][:] = table[toks[b]][:]
// ---------------------------------------------------------------------------
__global__ void embed_kernel(const int* __restrict__ toks,
                             const float* __restrict__ table,
                             float* __restrict__ tgt)
{
    int b = blockIdx.x, d = threadIdx.x;  // 320 threads
    tgt[b * ED + d] = table[(long long)toks[b] * ED + d];
}

// ---------------------------------------------------------------------------
// xPos tables (shared by all layers / both attentions: total = 256, k_off = 0).
// sin_k/cos_k: [256][40] with downscale; sin_q/cos_q: [40] at p = q_offset.
// ---------------------------------------------------------------------------
__global__ void xpos_tables_kernel(const int* __restrict__ qoff_p,
                                   float* __restrict__ sin_k, float* __restrict__ cos_k,
                                   float* __restrict__ sin_q, float* __restrict__ cos_q)
{
    int tid = threadIdx.x;
    const float hd = (float)HD;
    for (int idx = tid; idx < TKC * (HD / 2); idx += 256) {
        int p = idx / (HD / 2), i = idx - p * (HD / 2);
        float base  = (2.0f * i + 0.4f * hd) / (1.4f * hd);
        float power = (float)(p - TKC / 2) / 320.0f;     // (min_pos + p)/SCALE_BASE
        float scale = __powf(base, -power);              // downscale for K
        float invf  = __powf(10000.0f, -(float)i / (HD / 2));
        float s     = (float)p * invf;
        sin_k[idx] = __sinf(s) * scale;
        cos_k[idx] = __cosf(s) * scale;
    }
    if (tid < HD / 2) {
        int qoff   = qoff_p[0];
        int total  = qoff + 1;
        int minpos = -((total + 1) >> 1);
        int p      = total - 1;
        float base  = (2.0f * tid + 0.4f * hd) / (1.4f * hd);
        float power = (float)(minpos + p) / 320.0f;
        float scale = __powf(base, power);               // upscale for Q
        float invf  = __powf(10000.0f, -(float)tid / (HD / 2));
        float s     = (float)p * invf;
        sin_q[tid] = __sinf(s) * scale;
        cos_q[tid] = __cosf(s) * scale;
    }
}

// ---------------------------------------------------------------------------
// LayerNorm -> bf16. If cached != null: row r of [64*256] is
// concat(cached[b], tgt[b]) row; else row r of [64] is tgt[b].
// ---------------------------------------------------------------------------
__global__ __launch_bounds__(256) void layernorm_kernel(
    const float* __restrict__ cached,   // [64][255][320] or null
    const float* __restrict__ tgt,      // [64][320]
    const float* __restrict__ w, const float* __restrict__ b,
    unsigned short* __restrict__ out)
{
    int row = blockIdx.x;
    const float* src;
    if (cached) {
        int bb = row >> 8, t = row & 255;
        src = (t < TC) ? cached + (long long)(bb * TC + t) * ED
                       : tgt + (long long)bb * ED;
    } else {
        src = tgt + (long long)row * ED;
    }
    __shared__ float s1[256], s2[256];
    int tid = threadIdx.x;
    float a0 = src[tid];
    float a1 = (tid < ED - 256) ? src[256 + tid] : 0.0f;
    s1[tid] = a0 + a1;
    s2[tid] = a0 * a0 + a1 * a1;
    __syncthreads();
    for (int s = 128; s > 0; s >>= 1) {
        if (tid < s) { s1[tid] += s1[tid + s]; s2[tid] += s2[tid + s]; }
        __syncthreads();
    }
    float mu   = s1[0] * (1.0f / ED);
    float var  = s2[0] * (1.0f / ED) - mu * mu;
    float rstd = rsqrtf(var + 1e-5f);
    out[(long long)row * ED + tid] = f2bf((a0 - mu) * rstd * w[tid] + b[tid]);
    if (tid < ED - 256)
        out[(long long)row * ED + 256 + tid] =
            f2bf((a1 - mu) * rstd * w[256 + tid] + b[256 + tid]);
}

// ---------------------------------------------------------------------------
// Single-query attention per (hyp, head). Applies xPos to q and k on the fly,
// softmax over 256 keys in LDS, weighted V sum; writes bf16 output row.
// ---------------------------------------------------------------------------
__global__ __launch_bounds__(256) void attn_kernel(
    const float* __restrict__ Q,              // [64][320] f32 (bias applied)
    const unsigned short* __restrict__ Kb,    // [64][256][320] bf16
    const unsigned short* __restrict__ Vb,    // [64][256][320] bf16
    const float* __restrict__ sin_k, const float* __restrict__ cos_k, // [256][40]
    const float* __restrict__ sin_q, const float* __restrict__ cos_q, // [40]
    const unsigned char* __restrict__ mask,   // [64][256] or null
    unsigned short* __restrict__ AO)          // [64][320] bf16
{
    int b = blockIdx.x >> 2, h = blockIdx.x & 3;
    int tid = threadIdx.x;                    // = key index t
    __shared__ float sc[256];
    __shared__ float red[256];

    const float* q = Q + b * ED + h * HD;
    const unsigned short* krow = Kb + ((long long)(b * TKC + tid)) * ED + h * HD;
    float acc = 0.0f;
#pragma unroll 4
    for (int i = 0; i < HD / 2; ++i) {
        float q1 = q[2 * i], q2 = q[2 * i + 1];
        float cq = cos_q[i], sq = sin_q[i];
        float qr1 = q1 * cq - q2 * sq;
        float qr2 = q2 * cq + q1 * sq;
        float k1 = bf2f(krow[2 * i]), k2 = bf2f(krow[2 * i + 1]);
        float ck = cos_k[tid * (HD / 2) + i], sk = sin_k[tid * (HD / 2) + i];
        float kr1 = k1 * ck - k2 * sk;
        float kr2 = k2 * ck + k1 * sk;
        acc += qr1 * kr1 + qr2 * kr2;
    }
    acc *= 0.111803398874989485f;             // HEAD_DIM^-0.5
    if (mask && mask[b * TKC + tid]) acc = -__builtin_inff();
    sc[tid] = acc;
    red[tid] = acc;
    __syncthreads();
    for (int s = 128; s > 0; s >>= 1) {
        if (tid < s) red[tid] = fmaxf(red[tid], red[tid + s]);
        __syncthreads();
    }
    float mx = red[0];
    __syncthreads();
    float e = __expf(sc[tid] - mx);
    sc[tid] = e;
    red[tid] = e;
    __syncthreads();
    for (int s = 128; s > 0; s >>= 1) {
        if (tid < s) red[tid] += red[tid + s];
        __syncthreads();
    }
    float inv = 1.0f / red[0];
    if (tid < HD) {
        const unsigned short* vcol = Vb + (long long)b * TKC * ED + h * HD + tid;
        float o = 0.0f;
        for (int t = 0; t < TKC; ++t) o += sc[t] * bf2f(vcol[(long long)t * ED]);
        AO[b * ED + h * HD + tid] = f2bf(o * inv);
    }
}

// ---------------------------------------------------------------------------
extern "C" void kernel_launch(void* const* d_in, const int* in_sizes, int n_in,
                              void* d_out, int out_size, void* d_ws, size_t ws_size,
                              hipStream_t stream)
{
    const int*   last_toks = (const int*)  d_in[0];
    const float* cached    = (const float*)d_in[1];
    const float* memory    = (const float*)d_in[2];
    const unsigned char* memmask = (const unsigned char*)d_in[3];
    const int*   q_offset  = (const int*)  d_in[4];
    const float* embd      = (const float*)d_in[5];
    const float* ln_w      = (const float*)d_in[6];
    const float* ln_b      = (const float*)d_in[7];
    const float* sa_w      = (const float*)d_in[8];
    const float* sa_b      = (const float*)d_in[9];
    const float* ca_w      = (const float*)d_in[10];
    const float* ca_b      = (const float*)d_in[11];
    const float* ff1_w     = (const float*)d_in[12];
    const float* ff1_b     = (const float*)d_in[13];
    const float* ff2_w     = (const float*)d_in[14];
    const float* ff2_b     = (const float*)d_in[15];

    float* tgt = (float*)d_out;               // [64][320] — is the final output

    // ---- workspace bump allocator ----
    char* ws = (char*)d_ws;
    size_t off = 0;
    auto alloc = [&](size_t bytes) -> void* {
        void* p = ws + off;
        off += (bytes + 255) & ~(size_t)255;
        return p;
    };
    const size_t WSA  = (size_t)4 * ED * ED;         // 4 mats per attn block
    unsigned short* saT  = (unsigned short*)alloc(NL * WSA * 2);
    unsigned short* caT  = (unsigned short*)alloc(NL * WSA * 2);
    unsigned short* f1T  = (unsigned short*)alloc((size_t)NL * ED * FF * 2);
    unsigned short* f2T  = (unsigned short*)alloc((size_t)NL * FF * ED * 2);
    unsigned short* memb = (unsigned short*)alloc((size_t)NHYP * TKC * ED * 2);
    unsigned short* nc   = (unsigned short*)alloc((size_t)NHYP * TKC * ED * 2);
    unsigned short* Kb   = (unsigned short*)alloc((size_t)NHYP * TKC * ED * 2);
    unsigned short* Vb   = (unsigned short*)alloc((size_t)NHYP * TKC * ED * 2);
    float*          Qb   = (float*)alloc((size_t)NHYP * ED * 4);
    unsigned short* ntb  = (unsigned short*)alloc((size_t)NHYP * ED * 2);
    unsigned short* AOb  = (unsigned short*)alloc((size_t)NHYP * ED * 2);
    unsigned short* Hb   = (unsigned short*)alloc((size_t)NHYP * FF * 2);
    float* sink = (float*)alloc((size_t)TKC * (HD / 2) * 4);
    float* cosk = (float*)alloc((size_t)TKC * (HD / 2) * 4);
    float* sinq = (float*)alloc((size_t)(HD / 2) * 4);
    float* cosq = (float*)alloc((size_t)(HD / 2) * 4);

    auto gblocks = [](int M, int N) { int w = (M >> 4) * (N >> 6); return (w + 7) / 8; };

    // ---- one-time prep: weight transpose+bf16, memory bf16, xPos tables ----
    for (int l = 0; l < NL; ++l) {
        for (int m = 0; m < 4; ++m) {
            size_t o = ((size_t)l * 4 + m) * ED * ED;
            transconv_kernel<<<(ED/16)*(ED/16), 256, 0, stream>>>(sa_w + o, saT + o, ED, ED);
            transconv_kernel<<<(ED/16)*(ED/16), 256, 0, stream>>>(ca_w + o, caT + o, ED, ED);
        }
        transconv_kernel<<<(ED/16)*(FF/16), 256, 0, stream>>>(
            ff1_w + (size_t)l * ED * FF, f1T + (size_t)l * ED * FF, ED, FF);
        transconv_kernel<<<(FF/16)*(ED/16), 256, 0, stream>>>(
            ff2_w + (size_t)l * FF * ED, f2T + (size_t)l * FF * ED, FF, ED);
    }
    {
        long long n = (long long)NHYP * TKC * ED;
        f32_to_bf16_kernel<<<(unsigned)((n + 255) / 256), 256, 0, stream>>>(memory, memb, n);
    }
    xpos_tables_kernel<<<1, 256, 0, stream>>>(q_offset, sink, cosk, sinq, cosq);
    embed_kernel<<<NHYP, ED, 0, stream>>>(last_toks, embd, tgt);

    // ---- layers ----
    for (int l = 0; l < NL; ++l) {
        const float* cached_l = cached + (size_t)l * NHYP * TC * ED;
        const unsigned short* saT_l = saT + (size_t)l * WSA;
        const unsigned short* caT_l = caT + (size_t)l * WSA;
        const float* sab = sa_b + (size_t)l * 4 * ED;
        const float* cab = ca_b + (size_t)l * 4 * ED;
        const float* lw  = ln_w + (size_t)l * 3 * ED;
        const float* lb  = ln_b + (size_t)l * 3 * ED;

        // -- self attention --
        layernorm_kernel<<<NHYP * TKC, 256, 0, stream>>>(cached_l, tgt, lw, lb, nc);
        gemm_bf16_wmma<<<gblocks(NHYP*TKC, ED), 256, 0, stream>>>(
            nc, ED, saT_l + 1 * ED * ED, sab + 1 * ED, Kb, NHYP*TKC, ED, ED, 0, 0);
        gemm_bf16_wmma<<<gblocks(NHYP*TKC, ED), 256, 0, stream>>>(
            nc, ED, saT_l + 2 * ED * ED, sab + 2 * ED, Vb, NHYP*TKC, ED, ED, 0, 0);
        // Q from the last row of each hypothesis's nc block (== LN(tgt))
        gemm_bf16_wmma<<<gblocks(NHYP, ED), 256, 0, stream>>>(
            nc + (size_t)TC * ED, (long long)TKC * ED, saT_l, sab, Qb,
            NHYP, ED, ED, 1, 0);
        attn_kernel<<<NHYP * NH, 256, 0, stream>>>(
            Qb, Kb, Vb, sink, cosk, sinq, cosq, nullptr, AOb);
        gemm_bf16_wmma<<<gblocks(NHYP, ED), 256, 0, stream>>>(
            AOb, ED, saT_l + 3 * ED * ED, sab + 3 * ED, tgt, NHYP, ED, ED, 2, 0);

        // -- cross attention --
        layernorm_kernel<<<NHYP, 256, 0, stream>>>(nullptr, tgt, lw + ED, lb + ED, ntb);
        gemm_bf16_wmma<<<gblocks(NHYP*TKC, ED), 256, 0, stream>>>(
            memb, ED, caT_l + 1 * ED * ED, cab + 1 * ED, Kb, NHYP*TKC, ED, ED, 0, 0);
        gemm_bf16_wmma<<<gblocks(NHYP*TKC, ED), 256, 0, stream>>>(
            memb, ED, caT_l + 2 * ED * ED, cab + 2 * ED, Vb, NHYP*TKC, ED, ED, 0, 0);
        gemm_bf16_wmma<<<gblocks(NHYP, ED), 256, 0, stream>>>(
            ntb, ED, caT_l, cab, Qb, NHYP, ED, ED, 1, 0);
        attn_kernel<<<NHYP * NH, 256, 0, stream>>>(
            Qb, Kb, Vb, sink, cosk, sinq, cosq, memmask, AOb);
        gemm_bf16_wmma<<<gblocks(NHYP, ED), 256, 0, stream>>>(
            AOb, ED, caT_l + 3 * ED * ED, cab + 3 * ED, tgt, NHYP, ED, ED, 2, 0);

        // -- feed forward --
        layernorm_kernel<<<NHYP, 256, 0, stream>>>(nullptr, tgt, lw + 2*ED, lb + 2*ED, ntb);
        gemm_bf16_wmma<<<gblocks(NHYP, FF), 256, 0, stream>>>(
            ntb, ED, f1T + (size_t)l * ED * FF, ff1_b + (size_t)l * FF, Hb,
            NHYP, FF, ED, 0, 1);
        gemm_bf16_wmma<<<gblocks(NHYP, ED), 256, 0, stream>>>(
            Hb, FF, f2T + (size_t)l * FF * ED, ff2_b + (size_t)l * ED, tgt,
            NHYP, ED, FF, 2, 0);
    }
    (void)in_sizes; (void)n_in; (void)out_size; (void)ws_size;
}
